// SparseCoder_21397527069158
// MI455X (gfx1250) — matrix operand order repaired and checked by reading, not compiled, not run on hardware
//
#include <hip/hip_runtime.h>
#include <hip/hip_bf16.h>
#include <stdint.h>

// ---------------------------------------------------------------------------
// Problem constants
// ---------------------------------------------------------------------------
#define B_SZ   2048
#define D_INV  1024
#define L_SZ   32768
#define K_TOP  64
#define NCHUNK 8                       // L split into 8 chunks of 4096 cols
#define TILES_PER_CHUNK 256            // 4096 / 16
#define NTILE_TOTAL (L_SZ / 16)        // 2048 column tiles of 16
#define ROWTILES (B_SZ / 16)           // 128 row tiles of 16

// d_out layout (floats)
#define OUT_SAE  0
#define OUT_TOPA ((size_t)B_SZ * D_INV)                 // 2097152
#define OUT_TOPI (OUT_TOPA + (size_t)B_SZ * K_TOP)      // 2228224
#define OUT_FVU  (OUT_TOPI + (size_t)B_SZ * K_TOP)      // 2359296
#define OUT_AUX  (OUT_FVU + 1)

// workspace layout (bytes)
#define WS_BPACK    ((size_t)0)
#define WS_BPACK_SZ ((size_t)NTILE_TOTAL * 32 * 32 * 64)   // 128 MiB bf16 hi/lo packed W_enc
#define WS_CVAL     (WS_BPACK + WS_BPACK_SZ)
#define WS_CVAL_SZ  ((size_t)B_SZ * NCHUNK * K_TOP * 4)
#define WS_CIDX     (WS_CVAL + WS_CVAL_SZ)
#define WS_CIDX_SZ  WS_CVAL_SZ
#define WS_MEAN     (WS_CIDX + WS_CIDX_SZ)
#define WS_MEAN_SZ  ((size_t)D_INV * 4)
#define WS_PART     (WS_MEAN + WS_MEAN_SZ)

typedef __attribute__((ext_vector_type(16))) __bf16 v16bf;
typedef __attribute__((ext_vector_type(8)))  float  v8f;

// ---------------------------------------------------------------------------
// helpers
// ---------------------------------------------------------------------------
__device__ __forceinline__ v8f wmma_bf16(v16bf a, v16bf b, v8f c) {
  // D = A(16x32 bf16) * B(32x16 bf16) + C(16x16 f32)
  return __builtin_amdgcn_wmma_f32_16x16x32_bf16(
      /*neg_a=*/false, a, /*neg_b=*/false, b,
      /*c_mod=*/(short)0, c, /*reuse_a=*/false, /*reuse_b=*/false);
}

// compiler-only fence: stop reordering / register-caching of LDS accesses
// around cross-lane producer/consumer points (hardware keeps a wave's LDS
// operations in order, so no s_wait is needed for same-wave visibility).
__device__ __forceinline__ void lds_fence_wave() {
  __builtin_amdgcn_wave_barrier();
  asm volatile("" ::: "memory");
}

// wave-uniform broadcast from lane `src` (uniform) without __shfl's clamping
__device__ __forceinline__ float bcast_f32(float v, int src) {
  return __int_as_float(__builtin_amdgcn_ds_bpermute(src << 2, __float_as_int(v)));
}

// split 8 f32 into 16 packed bf16 (truncation): f = hi + residual, lo = bf16(residual)
__device__ __forceinline__ void split_pack8(const float* f, uint32_t* hi, uint32_t* lo) {
#pragma unroll
  for (int p = 0; p < 4; ++p) {
    uint32_t b0 = __float_as_uint(f[2 * p + 0]);
    uint32_t b1 = __float_as_uint(f[2 * p + 1]);
    uint32_t h0 = b0 & 0xFFFF0000u;
    uint32_t h1 = b1 & 0xFFFF0000u;
    hi[p] = (b0 >> 16) | h1;
    float r0 = f[2 * p + 0] - __uint_as_float(h0);
    float r1 = f[2 * p + 1] - __uint_as_float(h1);
    lo[p] = (__float_as_uint(r0) >> 16) | (__float_as_uint(r1) & 0xFFFF0000u);
  }
}

// ---------------------------------------------------------------------------
// Pass P: prepack encoder_w into WMMA-B lane layout, bf16 hi/lo split.
// B pack layout used here (16-bit, 32x16): lanes 0-15 hold K=0..15 of column n=lane,
// lanes 16-31 hold K=16..31 of column n=lane-16.  K contiguous per lane.
// Entry (t, kc, lane): 32B hi + 32B lo.
// ---------------------------------------------------------------------------
__global__ __launch_bounds__(256) void k_pack_w(const float* __restrict__ W,
                                                uint8_t* __restrict__ ws) {
  int gid = blockIdx.x * blockDim.x + threadIdx.x;
  int l  = gid & 31;
  int kc = (gid >> 5) & 31;
  int t  = gid >> 10;
  if (t >= NTILE_TOTAL) return;
  int n  = t * 16 + (l & 15);
  int k0 = kc * 32 + ((l & 16) ? 16 : 0);
  const float* row = W + (size_t)n * D_INV + k0;
  float f[16];
#pragma unroll
  for (int j = 0; j < 16; ++j) f[j] = row[j];
  uint32_t hi[8], lo[8];
  split_pack8(f, hi, lo);
  split_pack8(f + 8, hi + 4, lo + 4);
  uint4* dst = (uint4*)(ws + WS_BPACK + (((size_t)t * 32 + kc) * 32 + l) * 64);
  dst[0] = make_uint4(hi[0], hi[1], hi[2], hi[3]);
  dst[1] = make_uint4(hi[4], hi[5], hi[6], hi[7]);
  dst[2] = make_uint4(lo[0], lo[1], lo[2], lo[3]);
  dst[3] = make_uint4(lo[4], lo[5], lo[6], lo[7]);
}

// ---------------------------------------------------------------------------
// Pass M: per-feature mean of x (for total_variance)
// ---------------------------------------------------------------------------
__global__ __launch_bounds__(256) void k_colmean(const float* __restrict__ x,
                                                 float* __restrict__ mean) {
  int d = blockIdx.x * blockDim.x + threadIdx.x;
  if (d >= D_INV) return;
  float s = 0.f;
  for (int b = 0; b < B_SZ; ++b) s += x[(size_t)b * D_INV + d];
  mean[d] = s * (1.0f / (float)B_SZ);
}

// ---------------------------------------------------------------------------
// Pass G: fused encoder GEMM (WMMA bf16 split) + relu + per-chunk top-64.
// Block = 16 batch rows x 4096 latent cols. 8 waves; wave w owns col tiles
// {w, w+8, ...}.  A tile staged in LDS in WMMA-A layout; B streamed from the
// prepacked buffer.  Per-wave top-64 lists in LDS (thresholds in registers),
// block merge at the end.
// ---------------------------------------------------------------------------
__global__ __launch_bounds__(256) void k_encode_topk(
    const float* __restrict__ x, const float* __restrict__ enc_b,
    const float* __restrict__ b_dec, const uint8_t* __restrict__ ws_b,
    float* __restrict__ cand_val, int* __restrict__ cand_idx) {
  __shared__ uint32_t sA[2][32][32][8];   // [hi/lo][kc][lane][8 dwords] = 64 KiB
  __shared__ float    s_tval[8][16][64];  // per-wave per-row top lists   32 KiB
  __shared__ int      s_tidx[8][16][64];  //                              32 KiB

  const int tid  = threadIdx.x;
  const int w    = tid >> 5;
  const int lane = tid & 31;
  const int rowTile = blockIdx.x >> 3;
  const int chunk   = blockIdx.x & 7;

  // ---- stage A tile: xc = x - b_dec, split bf16 hi/lo, WMMA-A swizzle.
  // A layout used here (16-bit, 16x32): lane m holds row m; lanes 0-15: K=0..15;
  // lanes 16-31: K=16..31 (K contiguous per lane, matching B pack).
#pragma unroll
  for (int i = 0; i < 4; ++i) {
    int e  = tid + i * 256;     // entry id 0..1023 = (kc, lane)
    int kc = e >> 5;
    int l  = e & 31;
    int m  = l & 15;
    int base = (l & 16) ? 16 : 0;
    const float* xr = x + (size_t)(rowTile * 16 + m) * D_INV;
    int k0 = kc * 32 + base;
    float f[16];
#pragma unroll
    for (int j = 0; j < 8; ++j) f[j] = xr[k0 + j] - b_dec[k0 + j];
#pragma unroll
    for (int j = 0; j < 8; ++j) f[8 + j] = xr[k0 + 8 + j] - b_dec[k0 + 8 + j];
    uint32_t hi[8], lo[8];
    split_pack8(f, hi, lo);
    split_pack8(f + 8, hi + 4, lo + 4);
#pragma unroll
    for (int j = 0; j < 8; ++j) { sA[0][kc][l][j] = hi[j]; sA[1][kc][l][j] = lo[j]; }
  }
  // init top lists (acts >= 0, so -1 sentinels never beat real candidates)
  for (int i = tid; i < 8 * 16 * 64; i += 256) {
    (&s_tval[0][0][0])[i] = -1.0f;
    (&s_tidx[0][0][0])[i] = 0x7FFFFFFF;
  }
  __syncthreads();

  // per-row admission thresholds (wave-uniform, live in registers)
  float thr[16];
#pragma unroll
  for (int r = 0; r < 16; ++r) thr[r] = -1.0f;

  const int colTile0 = chunk * TILES_PER_CHUNK;
  for (int it = 0; it < 32; ++it) {
    const int t = colTile0 + w + (it << 3);
    const uint4* bp = (const uint4*)(ws_b + WS_BPACK + (size_t)t * 32 * 32 * 64);
    v8f acc = {0.f, 0.f, 0.f, 0.f, 0.f, 0.f, 0.f, 0.f};
#pragma unroll 2
    for (int kc = 0; kc < 32; ++kc) {
      const uint4* p = bp + (size_t)(kc * 32 + lane) * 4;
      __builtin_prefetch((const void*)(p + 128), 0, 3);  // next kc stream hint
      union { uint4 q[2]; v16bf v; } bh, bl;
      bh.q[0] = p[0]; bh.q[1] = p[1];
      bl.q[0] = p[2]; bl.q[1] = p[3];
      v16bf ah = *(const v16bf*)&sA[0][kc][lane][0];
      v16bf al = *(const v16bf*)&sA[1][kc][lane][0];
      acc = wmma_bf16(ah, bh.v, acc);   // hi*hi
      acc = wmma_bf16(ah, bl.v, acc);   // hi*lo
      acc = wmma_bf16(al, bh.v, acc);   // lo*hi
    }
    // epilogue: + encoder_b, relu.  C layout: lane = col (n & 15),
    // vgpr e = row e (lanes 0-15) / row e+8 (lanes 16-31).
    const float eb = enc_b[t * 16 + (lane & 15)];
    float vals[8];
#pragma unroll
    for (int e2 = 0; e2 < 8; ++e2) {
      float v = acc[e2] + eb;
      vals[e2] = v > 0.f ? v : 0.f;
    }
    const int colBase = t * 16;

    // ---- threshold-filtered insertion into this wave's per-row top-64
#pragma unroll
    for (int r = 0; r < 16; ++r) {
      float v = vals[r & 7];
      bool mine = ((lane >> 4) == (r >> 3));
      unsigned msk = (unsigned)__ballot(mine && (v > thr[r]));
      while (msk) {
        int src = __ffs(msk) - 1;
        msk &= msk - 1;
        float cv = bcast_f32(v, src);
        int   ci = colBase + (src & 15);
        if (cv <= thr[r]) continue;
        // wave-uniform min scan over the 64 entries (2 per lane, ds_load_b64)
        lds_fence_wave();
        float2 e01 = *(const float2*)&s_tval[w][r][lane * 2];
        float mv; int mp;
        if (e01.x <= e01.y) { mv = e01.x; mp = lane * 2; }
        else                { mv = e01.y; mp = lane * 2 + 1; }
#pragma unroll
        for (int off = 16; off; off >>= 1) {
          float ov = __shfl_xor(mv, off, 32);
          int   op = __shfl_xor(mp, off, 32);
          if (ov < mv || (ov == mv && op < mp)) { mv = ov; mp = op; }
        }
        if (cv > mv && lane == 0) {
          s_tval[w][r][mp] = cv;
          s_tidx[w][r][mp] = ci;
        }
        lds_fence_wave();
        thr[r] = mv;  // conservative lower bound on the new list-min
      }
    }
  }
  __syncthreads();

  // ---- merge 8 per-wave lists -> per-chunk top-64 (wave w: rows 2w, 2w+1)
  for (int rr = 0; rr < 2; ++rr) {
    const int r = w * 2 + rr;
    const int rowG = rowTile * 16 + r;
    for (int k = 0; k < K_TOP; ++k) {
      lds_fence_wave();
      float bv = -3.0e38f; int bi = 0x7FFFFFFF; int bc = 0;
#pragma unroll
      for (int s = 0; s < 16; ++s) {
        int c   = lane * 16 + s;            // 0..511 candidates
        int ww  = c >> 6, pos = c & 63;
        float vv = s_tval[ww][r][pos];
        int   ii = s_tidx[ww][r][pos];
        if (vv > bv || (vv == bv && ii < bi)) { bv = vv; bi = ii; bc = (ww * 16 + r) * 64 + pos; }
      }
#pragma unroll
      for (int off = 16; off; off >>= 1) {
        float ov = __shfl_xor(bv, off, 32);
        int   oi = __shfl_xor(bi, off, 32);
        int   oc = __shfl_xor(bc, off, 32);
        if (ov > bv || (ov == bv && oi < bi)) { bv = ov; bi = oi; bc = oc; }
      }
      if (lane == 0) {
        cand_val[((size_t)rowG * NCHUNK + chunk) * K_TOP + k] = bv;
        cand_idx[((size_t)rowG * NCHUNK + chunk) * K_TOP + k] = bi;
        (&s_tval[0][0][0])[bc] = -3.0e38f;
      }
    }
  }
}

// ---------------------------------------------------------------------------
// Pass T: merge 8 chunk candidate lists per row -> final sorted top-64
// ---------------------------------------------------------------------------
__global__ __launch_bounds__(256) void k_merge(const float* __restrict__ cand_val,
                                               const int* __restrict__ cand_idx,
                                               float* __restrict__ out) {
  __shared__ float mvS[8][512];
  __shared__ int   miS[8][512];
  const int tid = threadIdx.x;
#pragma unroll
  for (int j = 0; j < 16; ++j) {
    int e = tid + 256 * j;          // 0..4095
    int row = e >> 9, pos = e & 511;
    size_t g = (size_t)(blockIdx.x * 8 + row) * (NCHUNK * K_TOP) + pos;
    mvS[row][pos] = cand_val[g];
    miS[row][pos] = cand_idx[g];
  }
  __syncthreads();
  const int w = tid >> 5, lane = tid & 31;
  const int rowG = blockIdx.x * 8 + w;
  for (int k = 0; k < K_TOP; ++k) {
    lds_fence_wave();
    float bv = -3.0e38f; int bi = 0x7FFFFFFF; int bc = 0;
#pragma unroll
    for (int s = 0; s < 16; ++s) {
      int c = lane * 16 + s;
      float vv = mvS[w][c];
      int   ii = miS[w][c];
      if (vv > bv || (vv == bv && ii < bi)) { bv = vv; bi = ii; bc = c; }
    }
#pragma unroll
    for (int off = 16; off; off >>= 1) {
      float ov = __shfl_xor(bv, off, 32);
      int   oi = __shfl_xor(bi, off, 32);
      int   oc = __shfl_xor(bc, off, 32);
      if (ov > bv || (ov == bv && oi < bi)) { bv = ov; bi = oi; bc = oc; }
    }
    if (lane == 0) {
      out[OUT_TOPA + (size_t)rowG * K_TOP + k] = bv > 0.f ? bv : 0.f;
      out[OUT_TOPI + (size_t)rowG * K_TOP + k] = (float)(bi & 0x7FFF);
      mvS[w][bc] = -3.0e38f;
    }
  }
}

// ---------------------------------------------------------------------------
// Pass D: sparse decode + per-row error/variance partial sums
// ---------------------------------------------------------------------------
__global__ __launch_bounds__(256) void k_decode(
    const float* __restrict__ x, const float* __restrict__ W_dec,
    const float* __restrict__ b_dec, const float* __restrict__ colmean,
    float* __restrict__ out, float* __restrict__ partial) {
  __shared__ float sc[K_TOP];
  __shared__ int   si[K_TOP];
  __shared__ float red[256];
  const int b = blockIdx.x, tid = threadIdx.x;
  if (tid < K_TOP) {
    sc[tid] = out[OUT_TOPA + (size_t)b * K_TOP + tid];
    si[tid] = (int)out[OUT_TOPI + (size_t)b * K_TOP + tid];
  }
  __syncthreads();
  float4 acc = {0.f, 0.f, 0.f, 0.f};
  const float4* W4 = (const float4*)W_dec;
#pragma unroll 4
  for (int k = 0; k < K_TOP; ++k) {
    float s = sc[k];
    float4 wr = W4[(size_t)si[k] * (D_INV / 4) + tid];
    acc.x += s * wr.x; acc.y += s * wr.y; acc.z += s * wr.z; acc.w += s * wr.w;
  }
  float4 bd = ((const float4*)b_dec)[tid];
  float4 o  = {acc.x + bd.x, acc.y + bd.y, acc.z + bd.z, acc.w + bd.w};
  ((float4*)out)[(size_t)b * (D_INV / 4) + tid] = o;  // OUT_SAE base = 0
  float4 xv = ((const float4*)x)[(size_t)b * (D_INV / 4) + tid];
  float4 cm = ((const float4*)colmean)[tid];
  float ex = xv.x - o.x, ey = xv.y - o.y, ez = xv.z - o.z, ew = xv.w - o.w;
  float err = ex * ex + ey * ey + ez * ez + ew * ew;
  float vx = xv.x - cm.x, vy = xv.y - cm.y, vz = xv.z - cm.z, vw = xv.w - cm.w;
  float var = vx * vx + vy * vy + vz * vz + vw * vw;

  red[tid] = err; __syncthreads();
  for (int s2 = 128; s2 > 0; s2 >>= 1) {
    if (tid < s2) red[tid] += red[tid + s2];
    __syncthreads();
  }
  if (tid == 0) partial[(size_t)b * 2 + 0] = red[0];
  __syncthreads();
  red[tid] = var; __syncthreads();
  for (int s2 = 128; s2 > 0; s2 >>= 1) {
    if (tid < s2) red[tid] += red[tid + s2];
    __syncthreads();
  }
  if (tid == 0) partial[(size_t)b * 2 + 1] = red[0];
}

// ---------------------------------------------------------------------------
// Pass F: deterministic final reduction -> fvu, auxk
// ---------------------------------------------------------------------------
__global__ __launch_bounds__(256) void k_finalize(const float* __restrict__ partial,
                                                  float* __restrict__ out) {
  __shared__ float re[256], rv[256];
  const int tid = threadIdx.x;
  float se = 0.f, sv = 0.f;
  for (int i = tid; i < B_SZ; i += 256) {
    se += partial[(size_t)i * 2 + 0];
    sv += partial[(size_t)i * 2 + 1];
  }
  re[tid] = se; rv[tid] = sv; __syncthreads();
  for (int s = 128; s > 0; s >>= 1) {
    if (tid < s) { re[tid] += re[tid + s]; rv[tid] += rv[tid + s]; }
    __syncthreads();
  }
  if (tid == 0) {
    out[OUT_FVU] = re[0] / rv[0];
    out[OUT_AUX] = 0.0f;
  }
}

// ---------------------------------------------------------------------------
extern "C" void kernel_launch(void* const* d_in, const int* in_sizes, int n_in,
                              void* d_out, int out_size, void* d_ws, size_t ws_size,
                              hipStream_t stream) {
  const float* x     = (const float*)d_in[0];
  const float* enc_w = (const float*)d_in[1];
  const float* enc_b = (const float*)d_in[2];
  const float* W_dec = (const float*)d_in[3];
  const float* b_dec = (const float*)d_in[4];
  float* out = (float*)d_out;
  uint8_t* ws = (uint8_t*)d_ws;
  float* cand_val = (float*)(ws + WS_CVAL);
  int*   cand_idx = (int*)(ws + WS_CIDX);
  float* colmean  = (float*)(ws + WS_MEAN);
  float* partial  = (float*)(ws + WS_PART);

  k_pack_w   <<<(NTILE_TOTAL * 32 * 32) / 256, 256, 0, stream>>>(enc_w, ws);
  k_colmean  <<<D_INV / 256, 256, 0, stream>>>(x, colmean);
  k_encode_topk<<<ROWTILES * NCHUNK, 256, 0, stream>>>(x, enc_b, b_dec, ws,
                                                       cand_val, cand_idx);
  k_merge    <<<B_SZ / 8, 256, 0, stream>>>(cand_val, cand_idx, out);
  k_decode   <<<B_SZ, 256, 0, stream>>>(x, W_dec, b_dec, colmean, out, partial);
  k_finalize <<<1, 256, 0, stream>>>(partial, out);
}
